// SelectiveCopyModel_7679401525305
// MI455X (gfx1250) — matrix operand order repaired and checked by reading, not compiled, not run on hardware
//
#include <hip/hip_runtime.h>
#include <hip/hip_bf16.h>

typedef __attribute__((ext_vector_type(16))) _Float16 v16h;
typedef __attribute__((ext_vector_type(8)))  _Float16 v8h;
typedef __attribute__((ext_vector_type(8)))  float    v8f;

#define BATCH_     4
#define SEQ_       1024
#define ROWS_      (BATCH_*SEQ_)       // 4096
#define DMODEL_    512
#define DINNER_    1024
#define NSTATE_    32
#define HHEADS_    16
#define PHEAD_     64
#define CONVDIM_   1088                 // D_INNER + 2*N_STATE
#define KCONV_     4
#define DINPROJ_   2128                 // 2*D_INNER + 2*N_STATE + H_HEADS
#define LAYERS_    4
#define VOCAB_     64
#define EPS_       1e-5f
#define CHUNK_     64
#define NCHUNK_    (SEQ_/CHUNK_)        // 16
#define STATE_SZ_  (PHEAD_*NSTATE_)     // 2048 per (b,h)

// ---------------------------------------------------------------------------
// Embedding gather: h[r, d] = emb[tok[r], d]
// ---------------------------------------------------------------------------
__global__ void embed_kernel(const int* __restrict__ tok,
                             const float* __restrict__ emb,
                             float* __restrict__ h) {
    int idx = blockIdx.x * blockDim.x + threadIdx.x;   // ROWS_*DMODEL_
    int r = idx >> 9;                                  // /512
    int d = idx & 511;
    h[idx] = emb[tok[r] * DMODEL_ + d];
}

// ---------------------------------------------------------------------------
// Weight convert + transpose:  W[K][N] f32  ->  Wt[N][K] f16
// ---------------------------------------------------------------------------
__global__ void wcvt_kernel(const float* __restrict__ W,
                            _Float16* __restrict__ Wt, int K, int N) {
    int idx = blockIdx.x * blockDim.x + threadIdx.x;   // N*K
    int n = idx / K;
    int k = idx - n * K;
    Wt[idx] = (_Float16)W[(size_t)k * N + n];
}

// ---------------------------------------------------------------------------
// Plain f32 -> f16 convert
// ---------------------------------------------------------------------------
__global__ void cvt_f16_kernel(const float* __restrict__ x,
                               _Float16* __restrict__ out, int n) {
    int i = blockIdx.x * blockDim.x + threadIdx.x;
    if (i < n) out[i] = (_Float16)x[i];
}

// ---------------------------------------------------------------------------
// RMSNorm over D, output f16 (GEMM A-operand).  One 256-thread block per row.
// ---------------------------------------------------------------------------
template<int D>
__global__ void rmsnorm_f16_kernel(const float* __restrict__ x,
                                   const float* __restrict__ w,
                                   _Float16* __restrict__ out) {
    int row = blockIdx.x;
    const float* xr = x + (size_t)row * D;
    float ss = 0.f;
    for (int i = threadIdx.x; i < D; i += 256) { float v = xr[i]; ss += v * v; }
    __shared__ float red[8];
    __shared__ float sscale;
    for (int o = 16; o; o >>= 1) ss += __shfl_down(ss, o);
    if ((threadIdx.x & 31) == 0) red[threadIdx.x >> 5] = ss;
    __syncthreads();
    if (threadIdx.x == 0) {
        float t = 0.f;
        for (int i = 0; i < 8; ++i) t += red[i];
        sscale = rsqrtf(t / (float)D + EPS_);
    }
    __syncthreads();
    float sc = sscale;
    for (int i = threadIdx.x; i < D; i += 256)
        out[(size_t)row * D + i] = (_Float16)(xr[i] * sc * w[i]);
}

// ---------------------------------------------------------------------------
// Gated RMSNorm:  v = y * silu(z);  out = f16( v * rsqrt(mean(v^2)+eps) * gw )
// ---------------------------------------------------------------------------
__global__ void gated_norm_kernel(const float* __restrict__ y,
                                  const float* __restrict__ zx,
                                  const float* __restrict__ gw,
                                  _Float16* __restrict__ out) {
    int row = blockIdx.x;
    const float* yr = y  + (size_t)row * DINNER_;
    const float* zr = zx + (size_t)row * DINPROJ_;
    float v[4];
    float ss = 0.f;
#pragma unroll
    for (int j = 0; j < 4; ++j) {
        int i = threadIdx.x + j * 256;
        float z = zr[i];
        float s = z / (1.f + expf(-z));
        float val = yr[i] * s;
        v[j] = val;
        ss += val * val;
    }
    __shared__ float red[8];
    __shared__ float sscale;
    for (int o = 16; o; o >>= 1) ss += __shfl_down(ss, o);
    if ((threadIdx.x & 31) == 0) red[threadIdx.x >> 5] = ss;
    __syncthreads();
    if (threadIdx.x == 0) {
        float t = 0.f;
        for (int i = 0; i < 8; ++i) t += red[i];
        sscale = rsqrtf(t / (float)DINNER_ + EPS_);
    }
    __syncthreads();
    float sc = sscale;
#pragma unroll
    for (int j = 0; j < 4; ++j) {
        int i = threadIdx.x + j * 256;
        out[(size_t)row * DINNER_ + i] = (_Float16)(v[j] * sc * gw[i]);
    }
}

// ---------------------------------------------------------------------------
// Causal depthwise conv (K=4) + bias + SiLU.
// ---------------------------------------------------------------------------
__global__ void conv_silu_kernel(const float* __restrict__ zx,
                                 const float* __restrict__ cw,
                                 const float* __restrict__ cb,
                                 float* __restrict__ xc) {
    int idx = blockIdx.x * blockDim.x + threadIdx.x;   // ROWS_*CONVDIM_
    int r = idx / CONVDIM_;
    int c = idx - r * CONVDIM_;
    int b = r >> 10;
    int s = r & 1023;
    const float* col = zx + (size_t)(b << 10) * DINPROJ_ + (DINNER_ + c);
    float acc = cb[c];
#pragma unroll
    for (int j = 0; j < KCONV_; ++j) {
        int sj = s - (KCONV_ - 1) + j;
        float xv = (sj >= 0) ? col[(size_t)sj * DINPROJ_] : 0.f;
        acc += cw[c * KCONV_ + j] * xv;
    }
    xc[idx] = acc / (1.f + expf(-acc));   // silu
}

// ---------------------------------------------------------------------------
// SSD chunked scan, phase A: per-chunk local scan from zero state.
// Grid: B*H*NCHUNK blocks, 256 threads.  Thread t owns p = t>>2 and the
// 8 state columns n = (t&3)*8..+7 (flat state index = t*8 + i).
// Writes y (local part + D skip term), S_local, and the chunk dA-product.
// ---------------------------------------------------------------------------
__global__ void ssd_chunk_local_kernel(const float* __restrict__ xconv,
                                       const float* __restrict__ zx,
                                       const float* __restrict__ dt_bias,
                                       const float* __restrict__ A_log,
                                       const float* __restrict__ Dp,
                                       float* __restrict__ y,
                                       float* __restrict__ S_local,
                                       float* __restrict__ Pprod) {
    int blk = blockIdx.x;                 // b*256 + h*16 + c
    int c = blk & (NCHUNK_ - 1);
    int h = (blk >> 4) & (HHEADS_ - 1);
    int b = blk >> 8;
    int t = threadIdx.x;
    int p  = t >> 2;
    int ng = (t & 3) << 3;
    float A   = -expf(A_log[h]);
    float dtb = dt_bias[h];
    float Dv  = Dp[h];
    float hs[8];
#pragma unroll
    for (int i = 0; i < 8; ++i) hs[i] = 0.f;
    float pacc = 1.f;
    int row0 = b * SEQ_ + c * CHUNK_;
    for (int s = 0; s < CHUNK_; ++s) {
        int row = row0 + s;
        const float* xr = xconv + (size_t)row * CONVDIM_;
        float dtraw = zx[(size_t)row * DINPROJ_ + (DINNER_ + CONVDIM_) + h] + dtb;
        float dtv = (dtraw > 20.f) ? dtraw : log1pf(expf(dtraw));   // softplus
        float dA  = expf(dtv * A);
        pacc *= dA;
        float xv  = xr[h * PHEAD_ + p];
        float dtx = dtv * xv;
        float acc = 0.f;
#pragma unroll
        for (int i = 0; i < 8; ++i) {
            float Bn = xr[DINNER_ + ng + i];
            float Cn = xr[DINNER_ + NSTATE_ + ng + i];
            hs[i] = dA * hs[i] + dtx * Bn;
            acc += hs[i] * Cn;
        }
        acc += __shfl_xor(acc, 1);
        acc += __shfl_xor(acc, 2);
        if ((t & 3) == 0)
            y[(size_t)row * DINNER_ + h * PHEAD_ + p] = acc + xv * Dv;
    }
    float* Sl = S_local + (size_t)blk * STATE_SZ_ + t * 8;
#pragma unroll
    for (int i = 0; i < 8; ++i) Sl[i] = hs[i];
    if (t == 0) Pprod[blk] = pacc;
}

// ---------------------------------------------------------------------------
// Phase B: chunk-level recurrence over 16 chunks (per b,h):
//   S_start[c] = state at chunk entry;  state = P[c]*state + S_local[c].
// Grid: B*H = 64 blocks, 256 threads, 8 state elems per thread.
// ---------------------------------------------------------------------------
__global__ void ssd_chunk_scan_kernel(const float* __restrict__ S_local,
                                      const float* __restrict__ Pprod,
                                      float* __restrict__ S_start) {
    int bh = blockIdx.x;                  // b*16 + h
    int t = threadIdx.x;
    float s[8];
#pragma unroll
    for (int i = 0; i < 8; ++i) s[i] = 0.f;
    int base = bh * NCHUNK_;
    for (int c = 0; c < NCHUNK_; ++c) {
        size_t idx = (size_t)(base + c) * STATE_SZ_ + t * 8;
        float* Sst = S_start + idx;
        const float* Sl = S_local + idx;
        float P = Pprod[base + c];
#pragma unroll
        for (int i = 0; i < 8; ++i) {
            Sst[i] = s[i];
            s[i] = P * s[i] + Sl[i];
        }
    }
}

// ---------------------------------------------------------------------------
// Phase C: cross-chunk correction:
//   y[t] += (prod_{s=chunk_start..t} dA_s) * (C_t . S_start[chunk])
// Grid: B*H*NCHUNK blocks; chunk 0 has zero start state -> early exit.
// ---------------------------------------------------------------------------
__global__ void ssd_chunk_fix_kernel(const float* __restrict__ xconv,
                                     const float* __restrict__ zx,
                                     const float* __restrict__ dt_bias,
                                     const float* __restrict__ A_log,
                                     const float* __restrict__ S_start,
                                     float* __restrict__ y) {
    int blk = blockIdx.x;                 // b*256 + h*16 + c
    int c = blk & (NCHUNK_ - 1);
    if (c == 0) return;
    int h = (blk >> 4) & (HHEADS_ - 1);
    int b = blk >> 8;
    int t = threadIdx.x;
    int p  = t >> 2;
    int ng = (t & 3) << 3;
    float A   = -expf(A_log[h]);
    float dtb = dt_bias[h];
    float Sst[8];
    {
        const float* Sp = S_start + (size_t)blk * STATE_SZ_ + t * 8;
#pragma unroll
        for (int i = 0; i < 8; ++i) Sst[i] = Sp[i];
    }
    float prefix = 1.f;
    int row0 = b * SEQ_ + c * CHUNK_;
    for (int s = 0; s < CHUNK_; ++s) {
        int row = row0 + s;
        const float* xr = xconv + (size_t)row * CONVDIM_;
        float dtraw = zx[(size_t)row * DINPROJ_ + (DINNER_ + CONVDIM_) + h] + dtb;
        float dtv = (dtraw > 20.f) ? dtraw : log1pf(expf(dtraw));
        float dA  = expf(dtv * A);
        prefix *= dA;
        float acc = 0.f;
#pragma unroll
        for (int i = 0; i < 8; ++i)
            acc += Sst[i] * xr[DINNER_ + NSTATE_ + ng + i];   // C_t[n] * S_start
        acc += __shfl_xor(acc, 1);
        acc += __shfl_xor(acc, 2);
        if ((t & 3) == 0)
            y[(size_t)row * DINNER_ + h * PHEAD_ + p] += prefix * acc;
    }
}

// ---------------------------------------------------------------------------
// WMMA GEMM:  C[M][N] f32 = A[M][K] f16 x Bt[N][K] f16 (+ bias + resid)
// One wave computes a 64x16 output strip: 4 M-tiles share one B (weight)
// fragment -> 10 b128 loads per 4 v_wmma_f32_16x16x32_f16.
// M multiple of 64, N multiple of 16, K multiple of 32.
// Fragment layouts per CDNA5 ISA 7.12.2.
// ---------------------------------------------------------------------------
__global__ void gemm_wmma_kernel(const _Float16* __restrict__ A,
                                 const _Float16* __restrict__ Bt,
                                 const float* __restrict__ bias,
                                 const float* __restrict__ resid,
                                 float* __restrict__ C,
                                 int M, int N, int K) {
    int gwave = (blockIdx.x * blockDim.x + threadIdx.x) >> 5;
    int lane  = threadIdx.x & 31;
    int tilesN = N >> 4;
    int tMb = gwave / tilesN;             // 64-row M-block
    int tN  = gwave - tMb * tilesN;
    if (tMb >= (M >> 6)) return;
    int lh = lane & 15;
    int kh = lane >> 4;
    const _Float16* arow0 = A  + (size_t)((tMb << 6) + lh) * K;
    const _Float16* brow  = Bt + (size_t)((tN  << 4) + lh) * K;
    v8f acc[4];
#pragma unroll
    for (int mi = 0; mi < 4; ++mi)
        acc[mi] = (v8f){0.f, 0.f, 0.f, 0.f, 0.f, 0.f, 0.f, 0.f};
    for (int k = 0; k < K; k += 32) {
        v8h b0 = *(const v8h*)(brow + k + kh * 8);
        v8h b1 = *(const v8h*)(brow + k + 16 + kh * 8);
        v16h bv;
#pragma unroll
        for (int i = 0; i < 8; ++i) { bv[i] = b0[i]; bv[i + 8] = b1[i]; }
#pragma unroll
        for (int mi = 0; mi < 4; ++mi) {
            const _Float16* ar = arow0 + (size_t)(mi * 16) * K;
            v8h a0 = *(const v8h*)(ar + k + kh * 8);
            v8h a1 = *(const v8h*)(ar + k + 16 + kh * 8);
            v16h av;
#pragma unroll
            for (int i = 0; i < 8; ++i) { av[i] = a0[i]; av[i + 8] = a1[i]; }
            acc[mi] = __builtin_amdgcn_wmma_f32_16x16x32_f16(
                /*neg_a=*/false, av, /*neg_b=*/false, bv,
                /*c_mod=*/(short)0, acc[mi], /*reuse_a=*/false, /*reuse_b=*/false);
        }
    }
    int n = (tN << 4) + lh;
    float bn = bias ? bias[n] : 0.f;
#pragma unroll
    for (int mi = 0; mi < 4; ++mi) {
#pragma unroll
        for (int r = 0; r < 8; ++r) {
            int m = (tMb << 6) + mi * 16 + (kh << 3) + r;
            size_t o = (size_t)m * N + n;
            float v = acc[mi][r] + bn;
            if (resid) v += resid[o];
            C[o] = v;
        }
    }
}

// ---------------------------------------------------------------------------
// Host orchestration
// ---------------------------------------------------------------------------
extern "C" void kernel_launch(void* const* d_in, const int* in_sizes, int n_in,
                              void* d_out, int out_size, void* d_ws, size_t ws_size,
                              hipStream_t stream) {
    (void)in_sizes; (void)n_in; (void)out_size; (void)ws_size;

    const int*   tok     = (const int*)  d_in[0];
    const float* emb     = (const float*)d_in[1];
    const float* norm_w  = (const float*)d_in[2];
    const float* W_in    = (const float*)d_in[3];
    const float* b_in    = (const float*)d_in[4];
    const float* conv_w  = (const float*)d_in[5];
    const float* conv_b  = (const float*)d_in[6];
    const float* dt_bias = (const float*)d_in[7];
    const float* A_log   = (const float*)d_in[8];
    const float* D_par   = (const float*)d_in[9];
    const float* gnorm_w = (const float*)d_in[10];
    const float* W_out   = (const float*)d_in[11];
    const float* b_out   = (const float*)d_in[12];
    const float* head_w  = (const float*)d_in[13];
    const float* head_b  = (const float*)d_in[14];

    char* ws = (char*)d_ws;
    size_t off = 0;
    auto suballoc = [&](size_t bytes) -> char* {
        char* p = ws + off;
        off += (bytes + 255) & ~(size_t)255;
        return p;
    };
    float*    hbuf  = (float*)   suballoc((size_t)ROWS_ * DMODEL_  * 4);
    _Float16* a16   = (_Float16*)suballoc((size_t)ROWS_ * DINNER_  * 2);
    float*    zx    = (float*)   suballoc((size_t)ROWS_ * DINPROJ_ * 4);
    float*    xc    = (float*)   suballoc((size_t)ROWS_ * CONVDIM_ * 4);
    float*    ybuf  = (float*)   suballoc((size_t)ROWS_ * DINNER_  * 4);
    _Float16* wt16  = (_Float16*)suballoc((size_t)DINPROJ_ * DMODEL_ * 2);
    float*    Sloc  = (float*)   suballoc((size_t)BATCH_ * HHEADS_ * NCHUNK_ * STATE_SZ_ * 4);
    float*    Sstrt = (float*)   suballoc((size_t)BATCH_ * HHEADS_ * NCHUNK_ * STATE_SZ_ * 4);
    float*    Pprod = (float*)   suballoc((size_t)BATCH_ * HHEADS_ * NCHUNK_ * 4);

    embed_kernel<<<(ROWS_ * DMODEL_) / 256, 256, 0, stream>>>(tok, emb, hbuf);

    for (int l = 0; l < LAYERS_; ++l) {
        rmsnorm_f16_kernel<DMODEL_><<<ROWS_, 256, 0, stream>>>(
            hbuf, norm_w + l * DMODEL_, a16);
        wcvt_kernel<<<(DINPROJ_ * DMODEL_) / 256, 256, 0, stream>>>(
            W_in + (size_t)l * DMODEL_ * DINPROJ_, wt16, DMODEL_, DINPROJ_);
        // in-proj GEMM: waves = (4096/64)*(2128/16) = 64*133 = 8512 -> 1064 blocks
        gemm_wmma_kernel<<<(ROWS_ / 64) * (DINPROJ_ / 16) / 8, 256, 0, stream>>>(
            a16, wt16, b_in + l * DINPROJ_, nullptr, zx, ROWS_, DINPROJ_, DMODEL_);
        conv_silu_kernel<<<(ROWS_ * CONVDIM_) / 256, 256, 0, stream>>>(
            zx, conv_w + l * CONVDIM_ * KCONV_, conv_b + l * CONVDIM_, xc);
        // chunked SSD scan
        ssd_chunk_local_kernel<<<BATCH_ * HHEADS_ * NCHUNK_, 256, 0, stream>>>(
            xc, zx, dt_bias + l * HHEADS_, A_log + l * HHEADS_, D_par + l * HHEADS_,
            ybuf, Sloc, Pprod);
        ssd_chunk_scan_kernel<<<BATCH_ * HHEADS_, 256, 0, stream>>>(Sloc, Pprod, Sstrt);
        ssd_chunk_fix_kernel<<<BATCH_ * HHEADS_ * NCHUNK_, 256, 0, stream>>>(
            xc, zx, dt_bias + l * HHEADS_, A_log + l * HHEADS_, Sstrt, ybuf);
        gated_norm_kernel<<<ROWS_, 256, 0, stream>>>(
            ybuf, zx, gnorm_w + l * DINNER_, a16);
        wcvt_kernel<<<(DMODEL_ * DINNER_) / 256, 256, 0, stream>>>(
            W_out + (size_t)l * DINNER_ * DMODEL_, wt16, DINNER_, DMODEL_);
        // out-proj GEMM with fused residual: waves = 64*32 = 2048 -> 256 blocks
        gemm_wmma_kernel<<<(ROWS_ / 64) * (DMODEL_ / 16) / 8, 256, 0, stream>>>(
            a16, wt16, b_out + l * DMODEL_, hbuf, hbuf, ROWS_, DMODEL_, DINNER_);
    }

    // LM head
    cvt_f16_kernel<<<(ROWS_ * DMODEL_) / 256, 256, 0, stream>>>(hbuf, a16, ROWS_ * DMODEL_);
    wcvt_kernel<<<(VOCAB_ * DMODEL_) / 256, 256, 0, stream>>>(head_w, wt16, DMODEL_, VOCAB_);
    // waves = 64*4 = 256 -> 32 blocks
    gemm_wmma_kernel<<<(ROWS_ / 64) * (VOCAB_ / 16) / 8, 256, 0, stream>>>(
        a16, wt16, head_b, nullptr, (float*)d_out, ROWS_, VOCAB_, DMODEL_);
}
